// Wisard_68401649156855
// MI455X (gfx1250) — 8-bit matrix operand order repaired and checked by reading, compiled, not run on hardware
//
#include <hip/hip_runtime.h>
#include <stdint.h>

typedef __attribute__((ext_vector_type(8))) int v8i;

#define ENTRY_SIZE 1024
#define N_CLASSES  10
#define TUPLE_SIZE 16
#define N_RAMS     64
#define N_ADDR     65536
#define BATCH      8192
#define NCOL       (N_CLASSES * N_RAMS)


__global__ void zero_u32(uint32_t* __restrict__ p, int n) {
  int i = blockIdx.x * blockDim.x + threadIdx.x;
  if (i < n) p[i] = 0u;
}

__global__ void build_weights(const int* __restrict__ map,
                              uint8_t* __restrict__ wlo,
                              uint8_t* __restrict__ whi) {
  int tid = blockIdx.x * blockDim.x + threadIdx.x;
  if (tid >= N_CLASSES * ENTRY_SIZE) return;
  int c  = tid >> 10;
  int rt = tid & 1023;
  int r  = rt >> 4;
  int t  = rt & 15;
  int e  = map[tid];
  int n  = c * N_RAMS + r;
  if (t < 8) whi[(size_t)n * ENTRY_SIZE + e] = (uint8_t)(1u << (7  - t));
  else       wlo[(size_t)n * ENTRY_SIZE + e] = (uint8_t)(1u << (15 - t));
}

__global__ void pack_samples(const int* __restrict__ s, uint8_t* __restrict__ d, int n) {
  int i = blockIdx.x * blockDim.x + threadIdx.x;
  if (i < n) d[i] = (uint8_t)s[i];
}

__global__ __launch_bounds__(256)
void wisard_wmma(const uint8_t* __restrict__ A,
                 const uint8_t* __restrict__ Blo,
                 const uint8_t* __restrict__ Bhi,
                 const float*   __restrict__ table,
                 float*         __restrict__ out)
{
  const int wave = threadIdx.x >> 5;
  const int lane = threadIdx.x & 31;
  const int grp  = lane & 15;
  const int half = lane >> 4;

  const int mtile = blockIdx.x * 8 + wave;
  const int ntile = blockIdx.y;

  const uint8_t* arow = A   + (size_t)(mtile * 16 + grp) * ENTRY_SIZE;
  const uint8_t* blor = Blo + (size_t)(ntile * 16 + grp) * ENTRY_SIZE;
  const uint8_t* bhir = Bhi + (size_t)(ntile * 16 + grp) * ENTRY_SIZE;

  v8i acc_lo = {};
  v8i acc_hi = {};

  for (int k0 = 0; k0 < ENTRY_SIZE; k0 += 64) {
    v8i a, bl, bh;
#pragma unroll
    for (int i = 0; i < 8; ++i) {
      const int aoff = ((i >> 1) << 4) + ((i & 1) << 2) + (half << 3);
      a[i] = *(const int*)(arow + k0 + aoff);
    }
#pragma unroll
    for (int j = 0; j < 8; ++j) {
      const int boff = ((j >> 1) << 4) + ((j & 1) << 2) + (half << 3);
      bl[j] = *(const int*)(blor + k0 + boff);
      bh[j] = *(const int*)(bhir + k0 + boff);
    }
    acc_lo = __builtin_amdgcn_wmma_i32_16x16x64_iu8(false, a, false, bl, acc_lo, false, false);
    acc_hi = __builtin_amdgcn_wmma_i32_16x16x64_iu8(false, a, false, bh, acc_hi, false, false);
  }

  const int ncol = ntile * 16 + grp;
  const float* tbl = table + (size_t)ncol * N_ADDR;
  const int cls  = ncol >> 6;

#pragma unroll
  for (int v = 0; v < 8; ++v) {
    int addr  = acc_lo[v] + (acc_hi[v] << 8);
    float hit = tbl[addr];
    hit += __shfl_xor(hit, 1);
    hit += __shfl_xor(hit, 2);
    hit += __shfl_xor(hit, 4);
    hit += __shfl_xor(hit, 8);
    if (grp == 0) {
      int b = mtile * 16 + v + (half << 3);
      atomicAdd(&out[b * N_CLASSES + cls], hit);
    }
  }
}

extern "C" void kernel_launch(void* const* d_in, const int* in_sizes, int n_in,
                              void* d_out, int out_size, void* d_ws, size_t ws_size,
                              hipStream_t stream) {
  const int*   samples = (const int*)d_in[0];
  const int*   map     = (const int*)d_in[1];
  const float* table   = (const float*)d_in[2];
  float* out = (float*)d_out;

  uint8_t* ws  = (uint8_t*)d_ws;
  uint8_t* s8  = ws;
  uint8_t* wlo = ws + (size_t)BATCH * ENTRY_SIZE;
  uint8_t* whi = wlo + (size_t)NCOL * ENTRY_SIZE;

  const int wt_u32 = 2 * NCOL * ENTRY_SIZE / 4;
  zero_u32<<<(wt_u32 + 255) / 256, 256, 0, stream>>>((uint32_t*)wlo, wt_u32);
  zero_u32<<<(BATCH * N_CLASSES + 255) / 256, 256, 0, stream>>>((uint32_t*)out, BATCH * N_CLASSES);

  build_weights<<<(N_CLASSES * ENTRY_SIZE + 255) / 256, 256, 0, stream>>>(map, wlo, whi);

  const int ns = BATCH * ENTRY_SIZE;
  pack_samples<<<(ns + 255) / 256, 256, 0, stream>>>(samples, s8, ns);

  dim3 grid(BATCH / 16 / 8, NCOL / 16);
  wisard_wmma<<<grid, 256, 0, stream>>>(s8, wlo, whi, table, out);
}
